// TSPDecoder_57664230916581
// MI455X (gfx1250) — compile-verified, matrix operand order checked by reading, not run on hardware
//
#include <hip/hip_runtime.h>
#include <math.h>
#include <stdint.h>

#define BATCH 512
#define NNODE 128
#define HD    128
#define NHEAD 8
#define KDIM  16
#define BPB   16            // batch elements per decode workgroup (WMMA M dim)
#define C_SCALE 10.0f

typedef __attribute__((ext_vector_type(16))) __bf16 v16bf;
typedef __attribute__((ext_vector_type(8)))  __bf16 v8bf;
typedef __attribute__((ext_vector_type(8)))  float  v8f;
typedef __attribute__((ext_vector_type(4)))  unsigned int u32x4;
typedef __attribute__((ext_vector_type(4)))  int          i32x4;
typedef __attribute__((ext_vector_type(8)))  int          i32x8;

// ---- Tensor Data Mover availability / arity detection -----------------------
#if defined(__has_builtin)
#if __has_builtin(__builtin_amdgcn_tensor_load_to_lds) && \
    __has_builtin(__builtin_amdgcn_s_wait_tensorcnt)
#define HAVE_TDM 1
#endif
#endif
#if defined(__has_include)
#if __has_include(<hip/amd_detail/amd_gfx1250_TDM.h>)
#define TDM_ARITY6 1
#endif
#endif

#ifdef HAVE_TDM
// LDS byte offset of a __shared__ object (workgroup-relative, AS3 ptrtoint).
static __device__ __forceinline__ unsigned lds_off(const void* p) {
  return (unsigned)(uintptr_t)(__attribute__((address_space(3))) const void*)p;
}
// D# group 0: count=1 (+ optional gather flags in word0), lds addr, 57-bit
// global addr, descriptor type=2 ("image").
static __device__ __forceinline__ u32x4 tdm_g0(unsigned flags, unsigned lds,
                                               unsigned long long ga) {
  u32x4 g0;
  g0[0] = flags;
  g0[1] = lds;
  g0[2] = (unsigned)(ga & 0xffffffffull);
  g0[3] = (unsigned)((ga >> 32) & 0x01ffffffull) | 0x80000000u;  // type=2
  return g0;
}
// D# group 1 shared by both uses: data_size=4B, tensor_dim0=128,
// tensor_dim1=65536 (=B*N rows), tile_dim0=128, tile_dim1=16,
// tensor_dim0_stride=128.
static __device__ __forceinline__ i32x8 tdm_g1_16x128_f32() {
  i32x8 g1;
  g1[0] = 0x20000;                       // data_size=2 -> 4 bytes
  g1[1] = (int)(128u << 16);             // tensor_dim0[15:0]=128
  g1[2] = 0;                             // tensor_dim0[31:16]=0, dim1[15:0]=0
  g1[3] = (int)((128u << 16) | 1u);      // dim1[31:16]=1 (65536), tile_dim0=128
  g1[4] = 16;                            // tile_dim1 = 16 (rows / #indices)
  g1[5] = 128;                           // tensor_dim0_stride = 128
  g1[6] = 0;
  g1[7] = 0;
  return g1;
}
static __device__ __forceinline__ void tdm_issue(u32x4 g0, i32x8 g1,
                                                 i32x4 g2, i32x4 g3) {
#ifdef TDM_ARITY6
  i32x8 z;
#pragma unroll
  for (int k = 0; k < 8; ++k) z[k] = 0;
  __builtin_amdgcn_tensor_load_to_lds(g0, g1, g2, g3, z, 0);
#else
  __builtin_amdgcn_tensor_load_to_lds(g0, g1, g2, g3, 0);
#endif
}
#endif  // HAVE_TDM

static __device__ __forceinline__ __bf16 f2bf(float f) {
  unsigned u = __builtin_bit_cast(unsigned, f);
  unsigned r = (u + 0x7FFFu + ((u >> 16) & 1u)) >> 16;
  return __builtin_bit_cast(__bf16, (unsigned short)r);
}
static __device__ __forceinline__ float bfu2f(unsigned b16) {
  return __builtin_bit_cast(float, b16 << 16);
}
// K index held by fragment element j for lane-group g (16-bit A/B WMMA layout)
static __device__ __forceinline__ int kmap(int j, int g) {
  return (j < 8) ? (g * 8 + j) : (16 + g * 8 + (j - 8));
}
// A/B fragment rows are two contiguous 8-element runs: [g*8, +8) and [16+g*8, +8)
// -> exactly two 16B vector loads per lane.
static __device__ __forceinline__ v16bf load_frag_row(const __bf16* rowbase, int k0, int g) {
  v8bf lo = *(const v8bf*)(rowbase + k0 + g * 8);
  v8bf hi = *(const v8bf*)(rowbase + k0 + 16 + g * 8);
  return __builtin_shufflevector(lo, hi, 0, 1, 2, 3, 4, 5, 6, 7,
                                 8, 9, 10, 11, 12, 13, 14, 15);
}
static __device__ __forceinline__ float gumbel_noise(unsigned step, unsigned b, unsigned n) {
  unsigned h = step * 0x9E3779B9u ^ (b * 0x85EBCA6Bu) ^ (n * 0xC2B2AE35u);
  h ^= h >> 16; h *= 0x7FEB352Du; h ^= h >> 15; h *= 0x846CA68Bu; h ^= h >> 16;
  float u = (float)(h >> 8) * (1.0f / 16777216.0f);
  u = fminf(fmaxf(u, 1e-7f), 1.0f - 1e-7f);
  return -logf(-logf(u));
}

// ---------------------------------------------------------------------------
// Pack a K x 128 f32 weight into bf16 WMMA B-fragment order:
// tile(kt,nt) -> 32 lanes -> 16 contiguous bf16 per lane (one 32B load).
// ---------------------------------------------------------------------------
__global__ void pack_weight_kernel(const float* __restrict__ W,
                                   __bf16* __restrict__ P, int K) {
  int t = blockIdx.x * blockDim.x + threadIdx.x;
  if (t >= K * 128) return;
  int j    = t & 15;
  int lane = (t >> 4) & 31;
  int tile = t >> 9;
  int nt   = tile & 7;
  int kt   = tile >> 3;
  int g    = lane >> 4;
  int krow = kt * 32 + kmap(j, g);
  int ncol = nt * 16 + (lane & 15);
  P[t] = f2bf(W[krow * 128 + ncol]);
}

// h_avg[b][h] = mean over nodes of x[b][n][h]
__global__ void havg_kernel(const float* __restrict__ x, float* __restrict__ havg) {
  int t = blockIdx.x * blockDim.x + threadIdx.x;
  if (t >= BATCH * HD) return;
  int b = t >> 7, h = t & 127;
  float s = 0.f;
  for (int n = 0; n < NNODE; ++n) s += x[(((size_t)b * NNODE) + n) * HD + h];
  havg[t] = s * (1.0f / NNODE);
}

// ---------------------------------------------------------------------------
// Precompute GEMM: [B*N x 128] @ {Wk|Wv|Wkp} -> ks/vs/kproj (bf16), via WMMA.
// One block = 16 rows of the flattened x, staged into LDS by the TDM
// (tensor_load_to_lds, 2D 16x128 f32 tile); 8 waves cover 24 output tiles.
// ---------------------------------------------------------------------------
__global__ __launch_bounds__(256) void proj_wmma_kernel(
    const float* __restrict__ x,
    const __bf16* __restrict__ pWk, const __bf16* __restrict__ pWv,
    const __bf16* __restrict__ pWkp,
    __bf16* __restrict__ ksb, __bf16* __restrict__ vsb, __bf16* __restrict__ kpb) {
  __shared__ __bf16 xs[16][HD];
  const int tid  = threadIdx.x;
  const int lane = tid & 31;
  const int wid  = tid >> 5;
  const int g    = lane >> 4;
  const size_t r0 = (size_t)blockIdx.x * 16;

#ifdef HAVE_TDM
  __shared__ float xsF[16][HD];
  if (wid == 0) {
    i32x4 gz;
#pragma unroll
    for (int k = 0; k < 4; ++k) gz[k] = 0;
    u32x4 g0 = tdm_g0(/*count=1*/ 1u, lds_off(&xsF[0][0]),
                      (unsigned long long)(uintptr_t)(x + r0 * HD));
    tdm_issue(g0, tdm_g1_16x128_f32(), gz, gz);
    __builtin_amdgcn_s_wait_tensorcnt(0);
  }
  __syncthreads();
  for (int e = tid; e < 16 * HD; e += 256) {
    int r = e >> 7, c = e & 127;
    xs[r][c] = f2bf(xsF[r][c]);
  }
#else
  for (int e = tid; e < 16 * HD; e += 256) {
    int r = e >> 7, c = e & 127;
    xs[r][c] = f2bf(x[(r0 + r) * HD + c]);
  }
#endif
  __syncthreads();

  for (int job = wid; job < 24; job += 8) {
    int wsel = job >> 3;
    int nt   = job & 7;
    const __bf16* pk = (wsel == 0) ? pWk : (wsel == 1) ? pWv : pWkp;
    __bf16* dst      = (wsel == 0) ? ksb : (wsel == 1) ? vsb : kpb;
    v8f acc;
#pragma unroll
    for (int r = 0; r < 8; ++r) acc[r] = 0.f;
#pragma unroll
    for (int kk = 0; kk < 4; ++kk) {
      v16bf a = load_frag_row(&xs[lane & 15][0], kk * 32, g);
      v16bf bfr = *(const v16bf*)(pk + (((kk * 8 + nt) * 32 + lane) << 4));
      acc = __builtin_amdgcn_wmma_f32_16x16x32_bf16(false, a, false, bfr,
                                                    (short)0, acc, false, false);
    }
#pragma unroll
    for (int r = 0; r < 8; ++r)
      dst[(r0 + r + 8 * g) * HD + nt * 16 + (lane & 15)] = f2bf(acc[r]);
  }
}

// ---------------------------------------------------------------------------
// Persistent decode kernel: one block owns 16 batch elements for all 128
// sequential steps. Shared-weight projections are 16-row bf16 WMMA GEMMs with
// weight fragments pinned in VGPRs across the whole loop; per-batch attention
// score/softmax/mix is VALU (co-executes with the XDL WMMA pipe). The
// data-dependent state update x[b, idx_b, :] -> lastL uses the TDM gather
// mode (16 x 16-bit row indices, one tensor_load_to_lds per step).
// ---------------------------------------------------------------------------
__global__ __launch_bounds__(256) void tsp_decode_kernel(
    const float* __restrict__ x, const float* __restrict__ v_l,
    const float* __restrict__ v_f,
    const __bf16* __restrict__ ksb, const __bf16* __restrict__ vsb,
    const __bf16* __restrict__ kpb, const float* __restrict__ havg,
    const __bf16* __restrict__ pWq, const __bf16* __restrict__ pWo,
    float* __restrict__ out) {
  __shared__ float  haL[BPB][HD];
  __shared__ float  lastL[BPB][HD];
  __shared__ float  firstL[BPB][HD];
  __shared__ __bf16 hcL[BPB][3 * HD];
  __shared__ float  qsL[BPB][HD];
  __shared__ float  uL[BPB][NHEAD][NNODE];
  __shared__ __bf16 attB[BPB][HD];
  __shared__ __bf16 qB[BPB][HD];
  __shared__ float  logitsL[BPB][NNODE];
  __shared__ unsigned char maskL[BPB][NNODE];
  __shared__ int    selL[BPB];
  __shared__ float  logpL[BPB];

  const int tid  = threadIdx.x;
  const int lane = tid & 31;
  const int wid  = tid >> 5;
  const int g    = lane >> 4;
  const int b0   = blockIdx.x * BPB;
  const float inv_kd = 0.25f;                   // 1/sqrt(16)
  const float inv_h  = 0.088388347648318447f;   // 1/sqrt(128)

  // Loop-invariant weight fragments pinned in VGPRs: wave `wid` only ever
  // computes output tile nt == wid for both projections.
  v16bf wqf[12];
#pragma unroll
  for (int kk = 0; kk < 12; ++kk)
    wqf[kk] = *(const v16bf*)(pWq + (((kk * 8 + wid) * 32 + lane) << 4));
  v16bf wof[4];
#pragma unroll
  for (int kk = 0; kk < 4; ++kk)
    wof[kk] = *(const v16bf*)(pWo + (((kk * 8 + wid) * 32 + lane) << 4));

  for (int e = tid; e < BPB * HD; e += 256) {
    int b = e >> 7, h = e & 127;
    haL[b][h]    = havg[(b0 + b) * HD + h];
    lastL[b][h]  = v_l[h];
    firstL[b][h] = v_f[h];
    maskL[b][h]  = 0;  // NNODE == HD == 128
  }
  if (tid < BPB) { logpL[tid] = 0.f; selL[tid] = 0; }
  // Warm this block's K-side data toward the caches (global_prefetch_b8).
  for (int e = tid; e < BPB * NNODE; e += 256) {
    int b = e >> 7, n = e & 127;
    __builtin_prefetch(ksb + (((size_t)(b0 + b) * NNODE + n) * HD), 0, 1);
    __builtin_prefetch(kpb + (((size_t)(b0 + b) * NNODE + n) * HD), 0, 1);
  }
  __syncthreads();

  for (int i = 0; i < NNODE; ++i) {
    // --- 1) h_c = [h_avg | last | first] -> bf16 ---
    for (int e = tid; e < BPB * 3 * HD; e += 256) {
      int b = e / (3 * HD);
      int c = e - b * 3 * HD;
      float v = (c < HD) ? haL[b][c]
              : (c < 2 * HD) ? lastL[b][c - HD] : firstL[b][c - 2 * HD];
      hcL[b][c] = f2bf(v);
    }
    __syncthreads();

    // --- 2) qs = h_c @ Wq : 16x384 @ 384x128 WMMA, wave wid owns tile nt=wid ---
    {
      v8f acc;
#pragma unroll
      for (int r = 0; r < 8; ++r) acc[r] = 0.f;
#pragma unroll
      for (int kk = 0; kk < 12; ++kk) {
        v16bf a = load_frag_row(&hcL[lane & 15][0], kk * 32, g);
        acc = __builtin_amdgcn_wmma_f32_16x16x32_bf16(false, a, false, wqf[kk],
                                                      (short)0, acc, false, false);
      }
#pragma unroll
      for (int r = 0; r < 8; ++r)
        qsL[r + 8 * g][wid * 16 + (lane & 15)] = acc[r];
    }
    __syncthreads();

    // --- 3) u[b][h][n] = qs . ks * inv_kd, masked (per-batch: VALU) ---
    for (int e = tid; e < BPB * NHEAD * NNODE; e += 256) {
      int b = e >> 10;
      int h = (e >> 7) & 7;
      int n = e & 127;
      const uint4* kp =
          (const uint4*)(ksb + (((size_t)(b0 + b) * NNODE + n) * HD + h * KDIM));
      uint4 w0 = kp[0];
      uint4 w1 = kp[1];
      const float* q = &qsL[b][h * KDIM];
      float s = 0.f;
      s += q[0]  * bfu2f(w0.x & 0xffffu) + q[1]  * bfu2f(w0.x >> 16);
      s += q[2]  * bfu2f(w0.y & 0xffffu) + q[3]  * bfu2f(w0.y >> 16);
      s += q[4]  * bfu2f(w0.z & 0xffffu) + q[5]  * bfu2f(w0.z >> 16);
      s += q[6]  * bfu2f(w0.w & 0xffffu) + q[7]  * bfu2f(w0.w >> 16);
      s += q[8]  * bfu2f(w1.x & 0xffffu) + q[9]  * bfu2f(w1.x >> 16);
      s += q[10] * bfu2f(w1.y & 0xffffu) + q[11] * bfu2f(w1.y >> 16);
      s += q[12] * bfu2f(w1.z & 0xffffu) + q[13] * bfu2f(w1.z >> 16);
      s += q[14] * bfu2f(w1.w & 0xffffu) + q[15] * bfu2f(w1.w >> 16);
      s *= inv_kd;
      uL[b][h][n] = maskL[b][n] ? -1e30f : s;
    }
    __syncthreads();

    // --- 4) softmax over n + weighted sum with vs -> att (bf16) ---
    if (tid < BPB * NHEAD) {
      int b = tid >> 3, h = tid & 7;
      float m = -1e30f;
      for (int n = 0; n < NNODE; ++n) m = fmaxf(m, uL[b][h][n]);
      float sum = 0.f;
      for (int n = 0; n < NNODE; ++n) {
        float p = __expf(uL[b][h][n] - m);
        uL[b][h][n] = p;
        sum += p;
      }
      float attv[KDIM];
#pragma unroll
      for (int vd = 0; vd < KDIM; ++vd) attv[vd] = 0.f;
      for (int n = 0; n < NNODE; ++n) {
        float p = uL[b][h][n];
        const uint4* vp = (const uint4*)(
            vsb + (((size_t)(b0 + b) * NNODE + n) * HD + h * KDIM));
        uint4 w0 = vp[0];
        uint4 w1 = vp[1];
        attv[0]  += p * bfu2f(w0.x & 0xffffu); attv[1]  += p * bfu2f(w0.x >> 16);
        attv[2]  += p * bfu2f(w0.y & 0xffffu); attv[3]  += p * bfu2f(w0.y >> 16);
        attv[4]  += p * bfu2f(w0.z & 0xffffu); attv[5]  += p * bfu2f(w0.z >> 16);
        attv[6]  += p * bfu2f(w0.w & 0xffffu); attv[7]  += p * bfu2f(w0.w >> 16);
        attv[8]  += p * bfu2f(w1.x & 0xffffu); attv[9]  += p * bfu2f(w1.x >> 16);
        attv[10] += p * bfu2f(w1.y & 0xffffu); attv[11] += p * bfu2f(w1.y >> 16);
        attv[12] += p * bfu2f(w1.z & 0xffffu); attv[13] += p * bfu2f(w1.z >> 16);
        attv[14] += p * bfu2f(w1.w & 0xffffu); attv[15] += p * bfu2f(w1.w >> 16);
      }
      float inv = 1.f / sum;
#pragma unroll
      for (int vd = 0; vd < KDIM; ++vd)
        attB[b][h * KDIM + vd] = f2bf(attv[vd] * inv);
    }
    __syncthreads();

    // --- 5) q = att @ Wo : 16x128 @ 128x128 WMMA ---
    {
      v8f acc;
#pragma unroll
      for (int r = 0; r < 8; ++r) acc[r] = 0.f;
#pragma unroll
      for (int kk = 0; kk < 4; ++kk) {
        v16bf a = load_frag_row(&attB[lane & 15][0], kk * 32, g);
        acc = __builtin_amdgcn_wmma_f32_16x16x32_bf16(false, a, false, wof[kk],
                                                      (short)0, acc, false, false);
      }
#pragma unroll
      for (int r = 0; r < 8; ++r)
        qB[r + 8 * g][wid * 16 + (lane & 15)] = f2bf(acc[r]);
    }
    __syncthreads();

    // --- 6) logits = q . kproj^T via WMMA: kproj tile as A, q broadcast as B ---
    for (int job = wid; job < BPB * 8; job += 8) {
      int b  = job >> 3;
      int nt = job & 7;
      const __bf16* arow =
          kpb + (((size_t)(b0 + b) * NNODE) + nt * 16 + (lane & 15)) * HD;
      v8f acc;
#pragma unroll
      for (int r = 0; r < 8; ++r) acc[r] = 0.f;
#pragma unroll
      for (int kk = 0; kk < 4; ++kk) {
        v16bf a  = load_frag_row(arow, kk * 32, g);
        v16bf bb = load_frag_row(&qB[b][0], kk * 32, g);  // broadcast over N
        acc = __builtin_amdgcn_wmma_f32_16x16x32_bf16(false, a, false, bb,
                                                      (short)0, acc, false, false);
      }
      if ((lane & 15) == 0) {
#pragma unroll
        for (int r = 0; r < 8; ++r)
          logitsL[b][nt * 16 + r + 8 * g] = acc[r];
      }
    }
    __syncthreads();

    // --- 6b) tanh clip + mask ---
    for (int e = tid; e < BPB * NNODE; e += 256) {
      int b = e >> 7, n = e & 127;
      float lg = tanhf(logitsL[b][n] * inv_h) * C_SCALE;
      logitsL[b][n] = maskL[b][n] ? -1e30f : lg;
    }
    __syncthreads();

    // --- 7) gumbel-max sample + logp (one thread per batch element) ---
    if (tid < BPB) {
      int b = tid;
      float m = -1e30f;
      for (int n = 0; n < NNODE; ++n) m = fmaxf(m, logitsL[b][n]);
      float sum = 0.f, best = -1e30f;
      int bi = 0;
      for (int n = 0; n < NNODE; ++n) {
        float lg = logitsL[b][n];
        sum += __expf(lg - m);
        if (!maskL[b][n]) {
          float sc = lg + gumbel_noise((unsigned)i, (unsigned)(b0 + b), (unsigned)n);
          if (sc > best) { best = sc; bi = n; }
        }
      }
      logpL[b] += logitsL[b][bi] - m - logf(sum);
      selL[b] = bi;
      maskL[b][bi] = 1;
      out[(size_t)(b0 + b) * NNODE + i] = (float)bi;
    }
    __syncthreads();

    // --- 8) last/first state update: TDM gather of 16 selected x rows ---
#ifdef HAVE_TDM
    if (wid == 0) {
      unsigned idx[BPB];
#pragma unroll
      for (int b = 0; b < BPB; ++b)
        idx[b] = (unsigned)((b0 + b) * NNODE + selL[b]);  // < 65536: 16-bit ok
      i32x4 g2, g3;
#pragma unroll
      for (int k = 0; k < 4; ++k) {
        g2[k] = (int)(idx[2 * k]     | (idx[2 * k + 1] << 16));
        g3[k] = (int)(idx[8 + 2 * k] | (idx[9 + 2 * k] << 16));
      }
      // gather_mode=1 (bit31), 16-bit indices (bit30=0), count=1
      u32x4 g0 = tdm_g0(0x80000001u, lds_off(&lastL[0][0]),
                        (unsigned long long)(uintptr_t)x);
      tdm_issue(g0, tdm_g1_16x128_f32(), g2, g3);
      __builtin_amdgcn_s_wait_tensorcnt(0);
    }
    __syncthreads();
    if (i == 0) {
      for (int e = tid; e < BPB * HD; e += 256) {
        int b = e >> 7, h = e & 127;
        firstL[b][h] = lastL[b][h];
      }
      __syncthreads();
    }
#else
    for (int e = tid; e < BPB * HD; e += 256) {
      int b = e >> 7, h = e & 127;
      int bi = selL[b];
      float xv = x[(((size_t)(b0 + b) * NNODE) + bi) * HD + h];
      lastL[b][h] = xv;
      if (i == 0) firstL[b][h] = xv;
    }
    __syncthreads();
#endif
  }

  if (tid < BPB) out[(size_t)BATCH * NNODE + b0 + tid] = logpL[tid];
}

// ---------------------------------------------------------------------------
extern "C" void kernel_launch(void* const* d_in, const int* in_sizes, int n_in,
                              void* d_out, int out_size, void* d_ws, size_t ws_size,
                              hipStream_t stream) {
  (void)in_sizes; (void)n_in; (void)out_size; (void)ws_size;
  const float* x   = (const float*)d_in[0];
  const float* v_l = (const float*)d_in[1];
  const float* v_f = (const float*)d_in[2];
  const float* Wq  = (const float*)d_in[3];
  const float* Wk  = (const float*)d_in[4];
  const float* Wv  = (const float*)d_in[5];
  const float* Wo  = (const float*)d_in[6];
  const float* Wkp = (const float*)d_in[7];

  char* ws = (char*)d_ws;
  size_t off = 0;
  __bf16* ksb = (__bf16*)(ws + off); off += (size_t)BATCH * NNODE * HD * 2;
  __bf16* vsb = (__bf16*)(ws + off); off += (size_t)BATCH * NNODE * HD * 2;
  __bf16* kpb = (__bf16*)(ws + off); off += (size_t)BATCH * NNODE * HD * 2;
  float*  hav = (float*)(ws + off);  off += (size_t)BATCH * HD * 4;
  __bf16* pWq  = (__bf16*)(ws + off); off += (size_t)3 * HD * HD * 2;
  __bf16* pWk  = (__bf16*)(ws + off); off += (size_t)HD * HD * 2;
  __bf16* pWv  = (__bf16*)(ws + off); off += (size_t)HD * HD * 2;
  __bf16* pWo  = (__bf16*)(ws + off); off += (size_t)HD * HD * 2;
  __bf16* pWkp = (__bf16*)(ws + off); off += (size_t)HD * HD * 2;

  // Stage 0: pack weights into WMMA B-fragment order (bf16).
  pack_weight_kernel<<<(3 * HD * 128 + 255) / 256, 256, 0, stream>>>(Wq,  pWq,  3 * HD);
  pack_weight_kernel<<<(HD * 128 + 255) / 256,     256, 0, stream>>>(Wk,  pWk,  HD);
  pack_weight_kernel<<<(HD * 128 + 255) / 256,     256, 0, stream>>>(Wv,  pWv,  HD);
  pack_weight_kernel<<<(HD * 128 + 255) / 256,     256, 0, stream>>>(Wo,  pWo,  HD);
  pack_weight_kernel<<<(HD * 128 + 255) / 256,     256, 0, stream>>>(Wkp, pWkp, HD);

  // Stage 1: h_avg.
  havg_kernel<<<(BATCH * HD + 255) / 256, 256, 0, stream>>>(x, hav);

  // Stage 2: big WMMA GEMM -> ks, vs, kproj (bf16, L2 resident: ~50 MB).
  proj_wmma_kernel<<<(BATCH * NNODE) / 16, 256, 0, stream>>>(
      x, pWk, pWv, pWkp, ksb, vsb, kpb);

  // Stage 3: persistent sequential decode, 16 batch elements per block.
  tsp_decode_kernel<<<BATCH / BPB, 256, 0, stream>>>(
      x, v_l, v_f, ksb, vsb, kpb, hav, pWq, pWo, (float*)d_out);
}
